// GlobalAttentionPool_28518582846266
// MI455X (gfx1250) — matrix operand order repaired and checked by reading, not compiled, not run on hardware
//
#include <hip/hip_runtime.h>
#include <hip/hip_bf16.h>

#define N_NODES   65536
#define NUM_GRAPHS 512
#define MAX_N      192
#define DMODEL     512
#define NHEAD        8
#define DHEAD       64

typedef _Float16 v16h __attribute__((ext_vector_type(16)));
typedef _Float16 v8h  __attribute__((ext_vector_type(8)));
typedef float    v8f  __attribute__((ext_vector_type(8)));

__device__ __forceinline__ v8f zero_v8f() {
  v8f z;
#pragma unroll
  for (int i = 0; i < 8; ++i) z[i] = 0.0f;
  return z;
}

// CDNA5 WMMA: D(f32 16x16) = A(f16 16x32) * B(f16 32x16) + C
__device__ __forceinline__ v8f wmma_f16(v16h a, v16h b, v8f c) {
  return __builtin_amdgcn_wmma_f32_16x16x32_f16(
      /*neg_a=*/false, a, /*neg_b=*/false, b,
      /*c_mod=*/(short)0, c, /*reuse_a=*/false, /*reuse_b=*/false);
}

// A-fragment (16x32 f16) in contiguous k order (B uses the same):
//   lane: M = lane&15, khalf = lane>>4
//   elem e -> K = khalf*16 + e
// `base` points at row-start + k0 (element units).
__device__ __forceinline__ v16h load_a_frag(const _Float16* base, int khalf) {
  v8h lo = *(const v8h*)(base + khalf * 16);
  v8h hi = *(const v8h*)(base + khalf * 16 + 8);
  v16h r;
#pragma unroll
  for (int i = 0; i < 8; ++i) { r[i] = lo[i]; r[i + 8] = hi[i]; }
  return r;
}

// ---------------------------------------------------------------------------
// Kernel 1: counts/offsets of the sorted batch_indices (binary search).
// ---------------------------------------------------------------------------
__global__ void seg_offsets_kernel(const int* __restrict__ batch,
                                   int* __restrict__ counts,
                                   int* __restrict__ offsets) {
  int g = threadIdx.x;
  if (g >= NUM_GRAPHS) return;
  auto lb = [&](int v) {
    int lo = 0, hi = N_NODES;
    while (lo < hi) {
      int mid = (lo + hi) >> 1;
      if (batch[mid] < v) lo = mid + 1; else hi = mid;
    }
    return lo;
  };
  int o = lb(g);
  int e = lb(g + 1);
  offsets[g] = o;
  counts[g]  = e - o;
}

// ---------------------------------------------------------------------------
// Kernel 2: f32 -> f16 conversion (grid-stride).
// ---------------------------------------------------------------------------
__global__ void cvt_f16_kernel(const float* __restrict__ src,
                               _Float16* __restrict__ dst, int count) {
  int i = blockIdx.x * blockDim.x + threadIdx.x;
  int stride = gridDim.x * blockDim.x;
  for (; i < count; i += stride) dst[i] = (_Float16)src[i];
}

// ---------------------------------------------------------------------------
// Kernel 3: WMMA GEMM  Out = A @ W^T + bias   (then * scale), f16 out.
// A: [M, 512] f16 row-major, W: [512, 512] f16 row-major, M % 128 == 0.
// Block = 256 threads = 8 waves; block tile 128x128; wave tile 32x64
// (2 M-subtiles x 4 N-subtiles of 16x16); K-step 32.
// ---------------------------------------------------------------------------
__global__ void __launch_bounds__(256)
gemm_xwT_kernel(const _Float16* __restrict__ A, const _Float16* __restrict__ W,
                const float* __restrict__ bias, _Float16* __restrict__ Out,
                float scale) {
  const int lane  = threadIdx.x & 31;
  const int wave  = threadIdx.x >> 5;
  const int col   = lane & 15;
  const int khalf = lane >> 4;
  const int wm = wave & 3;   // M position within block tile
  const int wn = wave >> 2;  // N position within block tile
  const int rowBase = blockIdx.y * 128 + wm * 32;
  const int colBase = blockIdx.x * 128 + wn * 64;

  v8f acc[2][4];
#pragma unroll
  for (int mi = 0; mi < 2; ++mi)
#pragma unroll
    for (int ni = 0; ni < 4; ++ni) acc[mi][ni] = zero_v8f();

  for (int k0 = 0; k0 < DMODEL; k0 += 32) {
    v16h a[2], b[4];
#pragma unroll
    for (int mi = 0; mi < 2; ++mi)
      a[mi] = load_a_frag(A + (size_t)(rowBase + mi * 16 + col) * DMODEL + k0, khalf);
#pragma unroll
    for (int ni = 0; ni < 4; ++ni)
      // B elem e -> K = khalf*16+e, N = lane&15 ; B[k][n] = W[n][k]
      b[ni] = *(const v16h*)(W + (size_t)(colBase + ni * 16 + col) * DMODEL + k0 + khalf * 16);
#pragma unroll
    for (int mi = 0; mi < 2; ++mi)
#pragma unroll
      for (int ni = 0; ni < 4; ++ni)
        acc[mi][ni] = wmma_f16(a[mi], b[ni], acc[mi][ni]);
  }

#pragma unroll
  for (int mi = 0; mi < 2; ++mi)
#pragma unroll
    for (int ni = 0; ni < 4; ++ni) {
      const int c  = colBase + ni * 16 + col;
      const float bj = bias[c];
#pragma unroll
      for (int r = 0; r < 8; ++r) {
        const int row = rowBase + mi * 16 + r + 8 * khalf;
        Out[(size_t)row * DMODEL + c] = (_Float16)((acc[mi][ni][r] + bj) * scale);
      }
    }
}

// ---------------------------------------------------------------------------
// Kernel 4: build q_in rows (f16), packed per node, reference semantics:
//   q_in[i][d] = gqf[h*64 + d%64],  h = clip((t*512 + d) // (n*64), 0, 7)
// ---------------------------------------------------------------------------
__global__ void build_qin_kernel(const float* __restrict__ gqf,
                                 const int* __restrict__ counts,
                                 const int* __restrict__ offsets,
                                 const int* __restrict__ batch,
                                 _Float16* __restrict__ qin) {
  const int i = blockIdx.x;
  const int g = batch[i];
  const int n = max(counts[g], 1);
  const int t = i - offsets[g];
  const int denom = n * DHEAD;
  for (int d = threadIdx.x; d < DMODEL; d += blockDim.x) {
    int h = (t * DMODEL + d) / denom;
    h = min(h, NHEAD - 1);
    qin[(size_t)i * DMODEL + d] = (_Float16)gqf[h * DHEAD + (d & (DHEAD - 1))];
  }
}

// ---------------------------------------------------------------------------
// Kernel 5: per (graph, head) attention -> pooled softmax weights -> V sum.
// Block = 128 threads = 4 waves. Wave w processes query tiles w, w+4, w+8.
// Scores kept in registers (up to 16x192 f32 per wave); masked softmax with
// shfl reductions over the 16-lane groups of the WMMA C layout.
// pooled[g*512 + h*64 + dh] = (1/n) * sum_{t<n,s} softmax[t,s] * V[s, h*64+dh]
// ---------------------------------------------------------------------------
__global__ void __launch_bounds__(128)
attn_pool_kernel(const _Float16* __restrict__ Qh, const _Float16* __restrict__ Kh,
                 const _Float16* __restrict__ Vh, const int* __restrict__ counts,
                 const int* __restrict__ offsets, float* __restrict__ pooled) {
  const int h = blockIdx.x;
  const int g = blockIdx.y;
  const int tid = threadIdx.x;
  const int n = counts[g];
  float* poolg = pooled + (size_t)g * DMODEL + h * DHEAD;

  if (n == 0) {  // reference: mask all-false => pooled row is exactly zero
    if (tid < DHEAD) poolg[tid] = 0.0f;
    return;
  }
  const int ofs = offsets[g];

  __shared__ float wbarW[4][MAX_N];
  __shared__ float wbar[MAX_N];
  __shared__ float vred[128];

  const int lane = tid & 31, wave = tid >> 5;
  const int col = lane & 15, khalf = lane >> 4;

  for (int s = tid; s < 4 * MAX_N; s += 128) (&wbarW[0][0])[s] = 0.0f;
  __syncthreads();

  const int numT = (n + 15) >> 4;
  const int numS = numT;

  for (int tb = wave; tb < numT; tb += 4) {
    const int t0 = tb * 16;
    const int qrow = min(ofs + t0 + col, N_NODES - 1);
    const _Float16* qbase = Qh + (size_t)qrow * DMODEL + h * DHEAD;
    const v16h aq0 = load_a_frag(qbase, khalf);       // K 0..31 of head
    const v16h aq1 = load_a_frag(qbase + 32, khalf);  // K 32..63 of head

    v8f acc[12];
#pragma unroll
    for (int sb = 0; sb < 12; ++sb) acc[sb] = zero_v8f();

#pragma unroll
    for (int sb = 0; sb < 12; ++sb) {
      if (sb < numS) {
        const int srow = min(ofs + sb * 16 + col, N_NODES - 1);
        const _Float16* kbase = Kh + (size_t)srow * DMODEL + h * DHEAD;
        const v16h b0 = *(const v16h*)(kbase + khalf * 16);
        const v16h b1 = *(const v16h*)(kbase + 32 + khalf * 16);
        acc[sb] = wmma_f16(aq0, b0, acc[sb]);
        acc[sb] = wmma_f16(aq1, b1, acc[sb]);
      }
    }

    // ---- masked row softmax; C layout: row = t0 + r + 8*khalf, col s = sb*16 + (lane&15)
    float rmax[8];
#pragma unroll
    for (int r = 0; r < 8; ++r) rmax[r] = -3.0e38f;
#pragma unroll
    for (int sb = 0; sb < 12; ++sb) {
      if (sb < numS) {
        const int sc = sb * 16 + col;
#pragma unroll
        for (int r = 0; r < 8; ++r) {
          float v = (sc < n) ? acc[sb][r] : -1.0e9f;
          acc[sb][r] = v;
          rmax[r] = fmaxf(rmax[r], v);
        }
      }
    }
#pragma unroll
    for (int r = 0; r < 8; ++r)
      for (int m = 1; m < 16; m <<= 1)
        rmax[r] = fmaxf(rmax[r], __shfl_xor(rmax[r], m, 32));

    float rsum[8];
#pragma unroll
    for (int r = 0; r < 8; ++r) rsum[r] = 0.0f;
#pragma unroll
    for (int sb = 0; sb < 12; ++sb) {
      if (sb < numS) {
#pragma unroll
        for (int r = 0; r < 8; ++r) {
          float e = __expf(acc[sb][r] - rmax[r]);
          acc[sb][r] = e;
          rsum[r] += e;
        }
      }
    }
#pragma unroll
    for (int r = 0; r < 8; ++r)
      for (int m = 1; m < 16; m <<= 1)
        rsum[r] += __shfl_xor(rsum[r], m, 32);
    float rinv[8];
#pragma unroll
    for (int r = 0; r < 8; ++r) rinv[r] = 1.0f / rsum[r];

    // column sums of normalized weights over valid query rows (t < n)
#pragma unroll
    for (int sb = 0; sb < 12; ++sb) {
      if (sb < numS) {
        float cs = 0.0f;
#pragma unroll
        for (int r = 0; r < 8; ++r) {
          const int t = t0 + r + 8 * khalf;
          if (t < n) cs += acc[sb][r] * rinv[r];
        }
        cs += __shfl_xor(cs, 16, 32);  // merge the two M-halves
        if (khalf == 0) wbarW[wave][sb * 16 + col] += cs;
      }
    }
  }
  __syncthreads();

  for (int s = tid; s < MAX_N; s += 128)
    wbar[s] = wbarW[0][s] + wbarW[1][s] + wbarW[2][s] + wbarW[3][s];
  __syncthreads();

  // pooled_pre = (1/n) * sum_s wbar[s] * V[ofs+s, h*64+dh]
  const int dh = tid & 63, part = tid >> 6;
  float av = 0.0f;
  for (int s = part; s < n; s += 2)
    av += wbar[s] * (float)Vh[(size_t)(ofs + s) * DMODEL + h * DHEAD + dh];
  vred[tid] = av;
  __syncthreads();
  if (tid < DHEAD) poolg[tid] = (vred[tid] + vred[tid + 64]) / (float)n;
}

// ---------------------------------------------------------------------------
// Kernel 6: per-graph f32 tail: y = (n>0 ? pooled@w_o^T + b_o : 0);
//           out = relu(y @ w_p^T + b_p)
// ---------------------------------------------------------------------------
__global__ void __launch_bounds__(256)
final_proj_kernel(const float* __restrict__ pooled, const int* __restrict__ counts,
                  const float* __restrict__ w_o, const float* __restrict__ b_o,
                  const float* __restrict__ w_p, const float* __restrict__ b_p,
                  float* __restrict__ out) {
  const int g = blockIdx.x;
  const int tid = threadIdx.x;
  const int n = counts[g];
  __shared__ float sp[DMODEL];
  __shared__ float sy[DMODEL];

  for (int d = tid; d < DMODEL; d += 256) sp[d] = pooled[(size_t)g * DMODEL + d];
  __syncthreads();

  for (int j = tid; j < DMODEL; j += 256) {
    const float4* wr = (const float4*)(w_o + (size_t)j * DMODEL);
    float s = 0.0f;
    for (int k = 0; k < DMODEL / 4; ++k) {
      const float4 w4 = wr[k];
      s += sp[4 * k] * w4.x + sp[4 * k + 1] * w4.y +
           sp[4 * k + 2] * w4.z + sp[4 * k + 3] * w4.w;
    }
    sy[j] = (n > 0) ? (s + b_o[j]) : 0.0f;
  }
  __syncthreads();

  for (int j = tid; j < DMODEL; j += 256) {
    const float4* wr = (const float4*)(w_p + (size_t)j * DMODEL);
    float s = 0.0f;
    for (int k = 0; k < DMODEL / 4; ++k) {
      const float4 w4 = wr[k];
      s += sy[4 * k] * w4.x + sy[4 * k + 1] * w4.y +
           sy[4 * k + 2] * w4.z + sy[4 * k + 3] * w4.w;
    }
    out[(size_t)g * DMODEL + j] = fmaxf(s + b_p[j], 0.0f);
  }
}

// ---------------------------------------------------------------------------
extern "C" void kernel_launch(void* const* d_in, const int* in_sizes, int n_in,
                              void* d_out, int out_size, void* d_ws, size_t ws_size,
                              hipStream_t stream) {
  (void)in_sizes; (void)n_in; (void)out_size; (void)ws_size;

  const float* node_features = (const float*)d_in[0];
  const int*   batch         = (const int*)d_in[1];
  const float* gq            = (const float*)d_in[2];
  const float* w_q = (const float*)d_in[3];  const float* b_q = (const float*)d_in[4];
  const float* w_k = (const float*)d_in[5];  const float* b_k = (const float*)d_in[6];
  const float* w_v = (const float*)d_in[7];  const float* b_v = (const float*)d_in[8];
  const float* w_o = (const float*)d_in[9];  const float* b_o = (const float*)d_in[10];
  const float* w_p = (const float*)d_in[11]; const float* b_p = (const float*)d_in[12];
  float* out = (float*)d_out;

  char* ws = (char*)d_ws;
  size_t off = 0;
  auto wsalloc = [&](size_t bytes) -> void* {
    void* p = ws + off;
    off += (bytes + 255) & ~(size_t)255;
    return p;
  };
  int* counts  = (int*)wsalloc(NUM_GRAPHS * sizeof(int));
  int* offsets = (int*)wsalloc(NUM_GRAPHS * sizeof(int));
  _Float16* wqh = (_Float16*)wsalloc((size_t)DMODEL * DMODEL * 2);
  _Float16* wkh = (_Float16*)wsalloc((size_t)DMODEL * DMODEL * 2);
  _Float16* wvh = (_Float16*)wsalloc((size_t)DMODEL * DMODEL * 2);
  _Float16* B1  = (_Float16*)wsalloc((size_t)N_NODES * DMODEL * 2);  // Xh, then q_in
  _Float16* Kh  = (_Float16*)wsalloc((size_t)N_NODES * DMODEL * 2);
  _Float16* Vh  = (_Float16*)wsalloc((size_t)N_NODES * DMODEL * 2);
  _Float16* Qh  = (_Float16*)wsalloc((size_t)N_NODES * DMODEL * 2);
  float* pooled = (float*)wsalloc((size_t)NUM_GRAPHS * DMODEL * sizeof(float));

  seg_offsets_kernel<<<1, NUM_GRAPHS, 0, stream>>>(batch, counts, offsets);

  cvt_f16_kernel<<<4096, 256, 0, stream>>>(node_features, B1, N_NODES * DMODEL);
  cvt_f16_kernel<<<512, 256, 0, stream>>>(w_k, wkh, DMODEL * DMODEL);
  cvt_f16_kernel<<<512, 256, 0, stream>>>(w_v, wvh, DMODEL * DMODEL);
  cvt_f16_kernel<<<512, 256, 0, stream>>>(w_q, wqh, DMODEL * DMODEL);

  dim3 ggrid(DMODEL / 128, N_NODES / 128);
  gemm_xwT_kernel<<<ggrid, 256, 0, stream>>>(B1, wkh, b_k, Kh, 1.0f);
  gemm_xwT_kernel<<<ggrid, 256, 0, stream>>>(B1, wvh, b_v, Vh, 1.0f);

  build_qin_kernel<<<N_NODES, 256, 0, stream>>>(gq, counts, offsets, batch, B1);
  // fold score scale 1/sqrt(DH) = 0.125 into Q
  gemm_xwT_kernel<<<ggrid, 256, 0, stream>>>(B1, wqh, b_q, Qh, 0.125f);

  attn_pool_kernel<<<dim3(NHEAD, NUM_GRAPHS), 128, 0, stream>>>(Qh, Kh, Vh, counts,
                                                                offsets, pooled);
  final_proj_kernel<<<NUM_GRAPHS, 256, 0, stream>>>(pooled, counts, w_o, b_o,
                                                    w_p, b_p, out);
}
